// PointNet_Pre_SA_Module_55121610277169
// MI455X (gfx1250) — matrix operand order repaired and checked by reading, not compiled, not run on hardware
//
#include <hip/hip_runtime.h>
#include <hip/hip_bf16.h>

// ---- problem constants (from reference) ----
#define BB   32
#define MM   1024
#define KG   16
#define NIN  (MM + MM*KG)          // 17408 points per batch
#define RTOT (BB*MM*KG)            // 524288 rows of the per-point MLP
#define EPSB 1e-5f

typedef _Float16 f16;
typedef __attribute__((ext_vector_type(16))) _Float16 v16h;
typedef __attribute__((ext_vector_type(8)))  _Float16 v8h;
typedef __attribute__((ext_vector_type(8)))  float    v8f;
typedef __attribute__((ext_vector_type(4)))  int      v4i;

// async global->LDS copy path (ISA §10: GLOBAL_LOAD_ASYNC_TO_LDS_B128, ASYNCcnt)
#if defined(__has_builtin)
#if __has_builtin(__builtin_amdgcn_global_load_async_to_lds_b128) && \
    __has_builtin(__builtin_amdgcn_s_wait_asynccnt)
#define HAVE_ASYNC_LDS 1
#endif
#endif
#ifndef HAVE_ASYNC_LDS
#define HAVE_ASYNC_LDS 0
#endif

// ---------------------------------------------------------------------------
// WMMA fragment helpers (layouts per CDNA5 ISA 7.12.2)
// A: 16x32 f16, row M = lane%16. lanes 0-15: halves 0-7 => K 0-7, halves 8-15 => K 16-23
//                                 lanes 16-31: halves 0-7 => K 8-15, halves 8-15 => K 24-31
// works for global or LDS row base (addrspace inferred -> global_load_b128 / ds_load_b128)
// ---------------------------------------------------------------------------
__device__ __forceinline__ v16h load_afrag(const f16* __restrict__ rowbase, int lane, int kc) {
  int hi = (lane >> 4) << 3;                         // 0 or 8
  v8h lo = *(const v8h*)(rowbase + kc + hi);         // 16B aligned load
  v8h hh = *(const v8h*)(rowbase + kc + 16 + hi);
  v16h a;
#pragma unroll
  for (int j = 0; j < 8; ++j) { a[j] = lo[j]; a[j + 8] = hh[j]; }
  return a;
}

// B: 32x16 f16 (K x N). lane -> column N = nb + lane%16.
// lanes 0-15 hold K=kc+0..7,16..23 ; lanes 16-31 hold K=kc+8..15,24..31
// ldsW is row-major [O][64]  (W stored as (c_out, c_in), c_in = 64)
__device__ __forceinline__ v16h load_bfrag(const f16* __restrict__ ldsW, int lane, int nb, int kc) {
  const f16* wr = ldsW + (nb + (lane & 15)) * 64 + kc + ((lane >> 4) << 3);
  v16h b;
#pragma unroll
  for (int j = 0; j < 16; ++j) b[j] = wr[j < 8 ? j : j + 8];
  return b;
}

__device__ __forceinline__ v8f wmma_f16(v16h a, v16h b, v8f c) {
  return __builtin_amdgcn_wmma_f32_16x16x32_f16(false, a, false, b, (short)0, c, false, false);
}

// ---------------------------------------------------------------------------
// small utility kernels
// ---------------------------------------------------------------------------
__global__ void zero_stats_kernel(float* s) {
  int i = blockIdx.x * 256 + threadIdx.x;
  if (i < 1024) s[i] = 0.f;
}

__global__ void copy_xyz_kernel(const float* __restrict__ xyz, float* __restrict__ out) {
  int j = blockIdx.x * 256 + threadIdx.x;          // 32*1024*3 = 98304
  if (j < BB * MM * 3) {
    int b = j / (MM * 3), rem = j - b * (MM * 3);
    out[j] = xyz[(size_t)b * NIN * 3 + rem];
  }
}

// mean/var -> fused scale a = g*rstd, bias c = b - mean*g*rstd
__global__ void finalize_kernel(const float* __restrict__ sum, const float* __restrict__ sq,
                                const float* __restrict__ g, const float* __restrict__ bta,
                                float* __restrict__ a, float* __restrict__ c,
                                int nch, float inv_count) {
  int o = blockIdx.x * blockDim.x + threadIdx.x;
  if (o < nch) {
    float mean = sum[o] * inv_count;
    float var  = sq[o] * inv_count - mean * mean;
    float s    = g[o] * rsqrtf(var + EPSB);
    a[o] = s;
    c[o] = bta[o] - mean * s;
  }
}

// ---------------------------------------------------------------------------
// layer 0 (6->64) feature fetch + stats pass (scalar f32)
// ---------------------------------------------------------------------------
__device__ __forceinline__ void load_feats(const float* __restrict__ xyz,
                                           const float* __restrict__ pts,
                                           int r, float f[6]) {
  int b = r >> 14, i = r & 16383;                    // 16384 rows per batch
  size_t src = (size_t)b * NIN * 3 + (size_t)(MM + i) * 3;
  f[0] = xyz[src]; f[1] = xyz[src + 1]; f[2] = xyz[src + 2];
  f[3] = pts[src]; f[4] = pts[src + 1]; f[5] = pts[src + 2];
}

__global__ void __launch_bounds__(256) stats0_kernel(const float* __restrict__ xyz,
                                                     const float* __restrict__ pts,
                                                     const float* __restrict__ W0,
                                                     float* __restrict__ sum0,
                                                     float* __restrict__ sq0) {
  __shared__ float sW[64 * 6];
  __shared__ float lsum[64], lsq[64];
  int tid = threadIdx.x;
  for (int i = tid; i < 384; i += 256) sW[i] = W0[i];
  for (int i = tid; i < 64; i += 256) { lsum[i] = 0.f; lsq[i] = 0.f; }
  __syncthreads();

  int r = blockIdx.x * 256 + tid;
  float f[6]; load_feats(xyz, pts, r, f);
  int lane = tid & 31;
#pragma unroll 4
  for (int o = 0; o < 64; ++o) {
    const float* wr = &sW[o * 6];
    float y = f[0]*wr[0] + f[1]*wr[1] + f[2]*wr[2] + f[3]*wr[3] + f[4]*wr[4] + f[5]*wr[5];
    float s = y, q = y * y;
#pragma unroll
    for (int d = 16; d >= 1; d >>= 1) { s += __shfl_xor(s, d, 32); q += __shfl_xor(q, d, 32); }
    if (lane == 0) { atomicAdd(&lsum[o], s); atomicAdd(&lsq[o], q); }
  }
  __syncthreads();
  for (int i = tid; i < 64; i += 256) { atomicAdd(&sum0[i], lsum[i]); atomicAdd(&sq0[i], lsq[i]); }
}

// ---------------------------------------------------------------------------
// PASS A: build x1 = relu(BN(y0)) for 256 rows (scalar), stage in LDS,
//         store coalesced-vectorized to global, then WMMA y1 = x1*W1^T from LDS
//         and accumulate layer-1 per-channel sum/sumsq. Saves a full 64MB read.
// ---------------------------------------------------------------------------
__global__ void __launch_bounds__(256) fused_x1_stats1_kernel(const float* __restrict__ xyz,
                                                              const float* __restrict__ pts,
                                                              const float* __restrict__ W0,
                                                              const float* __restrict__ a0,
                                                              const float* __restrict__ c0,
                                                              const float* __restrict__ W1,
                                                              f16* __restrict__ xh,
                                                              float* __restrict__ sum1,
                                                              float* __restrict__ sq1) {
  __shared__ __align__(16) f16 sx[256 * 64];       // 32KB staged x1
  __shared__ __align__(16) f16 w1h[64 * 64];       // 8KB
  __shared__ float sW0[64 * 6];
  __shared__ float sA[64], sC[64];
  __shared__ float lsum[64], lsq[64];
  int tid = threadIdx.x;
  for (int i = tid; i < 384; i += 256) sW0[i] = W0[i];
  for (int i = tid; i < 64 * 64; i += 256) w1h[i] = (f16)W1[i];
  for (int i = tid; i < 64; i += 256) { sA[i] = a0[i]; sC[i] = c0[i]; lsum[i] = 0.f; lsq[i] = 0.f; }
  __syncthreads();

  // phase 1: one row per thread, fully unrolled so rowv stays in VGPRs
  int r = blockIdx.x * 256 + tid;
  float f[6]; load_feats(xyz, pts, r, f);
  f16 rowv[64];
#pragma unroll
  for (int o = 0; o < 64; ++o) {
    const float* wr = &sW0[o * 6];
    float y = f[0]*wr[0] + f[1]*wr[1] + f[2]*wr[2] + f[3]*wr[3] + f[4]*wr[4] + f[5]*wr[5];
    rowv[o] = (f16)fmaxf(0.f, sA[o] * y + sC[o]);
  }
  f16* lrow = &sx[tid * 64];
  f16* grow = xh + (size_t)r * 64;
#pragma unroll
  for (int vv = 0; vv < 8; ++vv) {
    v8h p;
#pragma unroll
    for (int j = 0; j < 8; ++j) p[j] = rowv[vv * 8 + j];
    *(v8h*)(lrow + vv * 8) = p;                    // ds_store_b128
    *(v8h*)(grow + vv * 8) = p;                    // global_store_b128
  }
  __syncthreads();

  // phase 2: 8 waves x 2 row-tiles of 16 => 256 rows; 4 col-tiles each
  int w = tid >> 5, lane = tid & 31;
  float sS[4] = {0.f, 0.f, 0.f, 0.f}, sQ[4] = {0.f, 0.f, 0.f, 0.f};
#pragma unroll
  for (int rt = 0; rt < 2; ++rt) {
    const f16* rowbase = &sx[(w * 32 + rt * 16 + (lane & 15)) * 64];
    v16h A0 = load_afrag(rowbase, lane, 0);
    v16h A1 = load_afrag(rowbase, lane, 32);
#pragma unroll
    for (int ct = 0; ct < 4; ++ct) {
      v8f acc = {};
      acc = wmma_f16(A0, load_bfrag(w1h, lane, ct * 16, 0), acc);
      acc = wmma_f16(A1, load_bfrag(w1h, lane, ct * 16, 32), acc);
#pragma unroll
      for (int v = 0; v < 8; ++v) { float y = acc[v]; sS[ct] += y; sQ[ct] += y * y; }
    }
  }
#pragma unroll
  for (int ct = 0; ct < 4; ++ct) {
    float s = sS[ct] + __shfl_xor(sS[ct], 16, 32);
    float q = sQ[ct] + __shfl_xor(sQ[ct], 16, 32);
    if (lane < 16) { atomicAdd(&lsum[ct * 16 + lane], s); atomicAdd(&lsq[ct * 16 + lane], q); }
  }
  __syncthreads();
  for (int i = tid; i < 64; i += 256) { atomicAdd(&sum1[i], lsum[i]); atomicAdd(&sq1[i], lsq[i]); }
}

// ---------------------------------------------------------------------------
// PASS B: stage 128 rows of x1 to LDS (async direct-to-LDS when available),
//         WMMA y1, BN+ReLU -> x2 in LDS, coalesced write-back of x2 (in place),
//         then WMMA y2 from LDS x2 and accumulate layer-2 stats.
// ---------------------------------------------------------------------------
__global__ void __launch_bounds__(256) fused_norm1_stats2_kernel(f16* __restrict__ xh,
                                                                 const float* __restrict__ W1,
                                                                 const float* __restrict__ a1,
                                                                 const float* __restrict__ c1,
                                                                 const float* __restrict__ W2,
                                                                 float* __restrict__ sum2,
                                                                 float* __restrict__ sq2) {
  __shared__ __align__(16) f16 sx1[128 * 64];      // 16KB
  __shared__ __align__(16) f16 sx2[128 * 64];      // 16KB
  __shared__ __align__(16) f16 w1h[64 * 64];       // 8KB
  __shared__ __align__(16) f16 w2h[128 * 64];      // 16KB
  __shared__ float sA[64], sC[64];
  int tid = threadIdx.x;
  size_t base = (size_t)blockIdx.x * 128 * 64;

#if HAVE_ASYNC_LDS
  // async global->LDS staging of x1 (GLOBAL_LOAD_ASYNC_TO_LDS_B128, ASYNCcnt);
  // overlaps with the f32->f16 weight conversion below.
  {
    v4i* gsrc = (v4i*)(xh + base);
    v4i* ldst = (v4i*)&sx1[0];
#pragma unroll
    for (int it = 0; it < 4; ++it) {
      int i = it * 256 + tid;                      // 1024 chunks of 16B
      __builtin_amdgcn_global_load_async_to_lds_b128(gsrc + i, ldst + i, 0, 0);
    }
  }
#else
  for (int i = tid; i < 1024; i += 256)
    *(v8h*)&sx1[i * 8] = *(const v8h*)(xh + base + (size_t)i * 8);
#endif
  for (int i = tid; i < 64 * 64; i += 256) w1h[i] = (f16)W1[i];
  for (int i = tid; i < 128 * 64; i += 256) w2h[i] = (f16)W2[i];
  for (int i = tid; i < 64; i += 256) { sA[i] = a1[i]; sC[i] = c1[i]; }
  // prefetch next block's x1 region (global_prefetch_b8)
  __builtin_prefetch(xh + base + 128 * 64 + (size_t)tid * 32, 0, 1);
#if HAVE_ASYNC_LDS
  __builtin_amdgcn_s_wait_asynccnt(0);
#endif
  __syncthreads();

  // phase 1: wave w -> rows [w*16, w*16+16), all 64 cols; A read before any write
  int w = tid >> 5, lane = tid & 31;
  {
    const f16* rowbase = &sx1[(w * 16 + (lane & 15)) * 64];
    v16h A0 = load_afrag(rowbase, lane, 0);
    v16h A1 = load_afrag(rowbase, lane, 32);
    int rbase = w * 16 + ((lane >> 4) << 3);
#pragma unroll
    for (int ct = 0; ct < 4; ++ct) {
      v8f acc = {};
      acc = wmma_f16(A0, load_bfrag(w1h, lane, ct * 16, 0), acc);
      acc = wmma_f16(A1, load_bfrag(w1h, lane, ct * 16, 32), acc);
      int N = ct * 16 + (lane & 15);
      float sc = sA[N], bi = sC[N];
#pragma unroll
      for (int v = 0; v < 8; ++v)
        sx2[(rbase + v) * 64 + N] = (f16)fmaxf(0.f, sc * acc[v] + bi);
    }
  }
  __syncthreads();

  // coalesced write-back of x2 (in place over x1's rows; all x1 reads done)
  for (int i = tid; i < 1024; i += 256)
    *(v8h*)(xh + base + (size_t)i * 8) = *(const v8h*)&sx2[i * 8];

  // phase 2: wave w -> col-tile w of W2 (128 outs), loop 8 row-tiles
  v16h Bf0 = load_bfrag(w2h, lane, w * 16, 0);
  v16h Bf1 = load_bfrag(w2h, lane, w * 16, 32);
  float sS = 0.f, sQ = 0.f;
#pragma unroll
  for (int rt = 0; rt < 8; ++rt) {
    const f16* rb2 = &sx2[(rt * 16 + (lane & 15)) * 64];
    v16h A0 = load_afrag(rb2, lane, 0);
    v16h A1 = load_afrag(rb2, lane, 32);
    v8f acc = {};
    acc = wmma_f16(A0, Bf0, acc);
    acc = wmma_f16(A1, Bf1, acc);
#pragma unroll
    for (int v = 0; v < 8; ++v) { float y = acc[v]; sS += y; sQ += y * y; }
  }
  sS += __shfl_xor(sS, 16, 32);
  sQ += __shfl_xor(sQ, 16, 32);
  if (lane < 16) { atomicAdd(&sum2[w * 16 + lane], sS); atomicAdd(&sq2[w * 16 + lane], sQ); }
}

// ---------------------------------------------------------------------------
// PASS C: y2 = x2*W2^T, BN+ReLU, max over K=16 (one 16-row tile == one (b,m) group)
// ---------------------------------------------------------------------------
__global__ void __launch_bounds__(256) gemm_final_kernel(const f16* __restrict__ xh,
                                                         const float* __restrict__ W,
                                                         const float* __restrict__ a2,
                                                         const float* __restrict__ c2,
                                                         float* __restrict__ out_pts) {
  __shared__ __align__(16) f16 ldsW[128 * 64];
  __shared__ float sA[128], sC[128];
  int tid = threadIdx.x;
  for (int i = tid; i < 128 * 64; i += 256) ldsW[i] = (f16)W[i];
  for (int i = tid; i < 128; i += 256) { sA[i] = a2[i]; sC[i] = c2[i]; }
  // prefetch next block's x2 rows
  __builtin_prefetch(xh + ((size_t)blockIdx.x + 1) * 128 * 64 + (size_t)tid * 32, 0, 1);
  __syncthreads();

  int w = tid >> 5, lane = tid & 31;
  int g  = blockIdx.x * 8 + w;                  // group = b*1024 + m ; 32768 groups
  int r0 = g * 16;
  const f16* rowbase = xh + (size_t)(r0 + (lane & 15)) * 64;
  v16h A0 = load_afrag(rowbase, lane, 0);
  v16h A1 = load_afrag(rowbase, lane, 32);
#pragma unroll
  for (int ct = 0; ct < 8; ++ct) {
    v8f acc = {};
    acc = wmma_f16(A0, load_bfrag(ldsW, lane, ct * 16, 0), acc);
    acc = wmma_f16(A1, load_bfrag(ldsW, lane, ct * 16, 32), acc);
    int N = ct * 16 + (lane & 15);
    float sc = sA[N], bi = sC[N];
    float mx = 0.f;                              // relu output is >= 0
#pragma unroll
    for (int v = 0; v < 8; ++v) mx = fmaxf(mx, fmaxf(0.f, sc * acc[v] + bi));
    mx = fmaxf(mx, __shfl_xor(mx, 16, 32));
    if (lane < 16) out_pts[(size_t)g * 128 + N] = mx;
  }
}

// ---------------------------------------------------------------------------
extern "C" void kernel_launch(void* const* d_in, const int* in_sizes, int n_in,
                              void* d_out, int out_size, void* d_ws, size_t ws_size,
                              hipStream_t stream) {
  (void)in_sizes; (void)n_in; (void)out_size; (void)ws_size;
  const float* xyz = (const float*)d_in[0];
  const float* pts = (const float*)d_in[1];
  const float* W0  = (const float*)d_in[2];
  const float* g0  = (const float*)d_in[3];
  const float* bb0 = (const float*)d_in[4];
  const float* W1  = (const float*)d_in[5];
  const float* g1  = (const float*)d_in[6];
  const float* bb1 = (const float*)d_in[7];
  const float* W2  = (const float*)d_in[8];
  const float* g2  = (const float*)d_in[9];
  const float* bb2 = (const float*)d_in[10];

  float* out_xyz = (float*)d_out;                 // 32*1024*3
  float* out_pts = out_xyz + BB * MM * 3;         // 32*1024*128

  // workspace: RTOT x 64 f16 activation buffer (64 MB, reused in place) + stats
  f16*   xh    = (f16*)d_ws;
  float* stats = (float*)((char*)d_ws + (size_t)RTOT * 64 * sizeof(f16));
  float *sum0 = stats + 0,   *sq0 = stats + 64,  *a0 = stats + 128, *c0 = stats + 192;
  float *sum1 = stats + 256, *sq1 = stats + 320, *a1 = stats + 384, *c1 = stats + 448;
  float *sum2 = stats + 512, *sq2 = stats + 640, *a2 = stats + 768, *c2 = stats + 896;
  const float invcnt = 1.0f / (float)RTOT;

  zero_stats_kernel<<<4, 256, 0, stream>>>(stats);
  copy_xyz_kernel<<<(BB * MM * 3 + 255) / 256, 256, 0, stream>>>(xyz, out_xyz);

  stats0_kernel<<<RTOT / 256, 256, 0, stream>>>(xyz, pts, W0, sum0, sq0);
  finalize_kernel<<<1, 128, 0, stream>>>(sum0, sq0, g0, bb0, a0, c0, 64, invcnt);

  fused_x1_stats1_kernel<<<RTOT / 256, 256, 0, stream>>>(xyz, pts, W0, a0, c0, W1,
                                                         xh, sum1, sq1);
  finalize_kernel<<<1, 128, 0, stream>>>(sum1, sq1, g1, bb1, a1, c1, 64, invcnt);

  fused_norm1_stats2_kernel<<<RTOT / 128, 256, 0, stream>>>(xh, W1, a1, c1, W2, sum2, sq2);
  finalize_kernel<<<1, 128, 0, stream>>>(sum2, sq2, g2, bb2, a2, c2, 128, invcnt);

  gemm_final_kernel<<<RTOT / 16 / 8, 256, 0, stream>>>(xh, W2, a2, c2, out_pts);
}
